// Seq2Seq_GRU_63127429316976
// MI455X (gfx1250) — matrix operand order repaired and checked by reading, not compiled, not run on hardware
//
#include <hip/hip_runtime.h>

#define HID   1024
#define INP   256
#define BATCH 64
#define TENC  256
#define TDEC  64
#define G3    3072   // 3*HID

#define GROUP 4                       // K-chunks staged per LDS buffer
#define ABUF  (4 * GROUP * 512)       // 8192 ushorts: A-frags, 4 mtiles
#define BBUF  (6 * GROUP * 512)       // 12288 ushorts: B-frags, 3 gates x 2 ntiles
#define BUFSZ (ABUF + BBUF)           // 20480 ushorts = 40KB per buffer

typedef __attribute__((ext_vector_type(16))) __bf16 v16bf;
typedef __attribute__((ext_vector_type(8)))  float  v8f;

__device__ __forceinline__ unsigned short f2bf(float f) {
  unsigned int u = __float_as_uint(f);
  u += 0x7fffu + ((u >> 16) & 1u);   // round-to-nearest-even
  return (unsigned short)(u >> 16);
}
__device__ __forceinline__ float sigm(float x) { return 1.0f / (1.0f + __expf(-x)); }

// CDNA5 async DMA: global -> LDS, 16B per issue, tracked by ASYNCcnt.
// The LDS address VGPR is derived from the real shared-memory pointer
// (low 32 bits of a shared-space generic address == DS byte offset), so the
// shared buffer escapes into the asm and the compiler must keep the
// subsequent ds_loads (the "memory" clobber then orders them).
__device__ __forceinline__ void async_g2l_b128(void* ldsptr, const void* gptr) {
  unsigned lds_off = (unsigned)(unsigned long long)ldsptr;
  asm volatile("global_load_async_to_lds_b128 %0, %1, off"
               :: "v"(lds_off), "v"(gptr) : "memory");
}
__device__ __forceinline__ void wait_async0() {
  asm volatile("s_wait_asynccnt 0" ::: "memory");
}

// ---------------------------------------------------------------------------
// Weight [N,K] f32 (row-major, pred = x @ W.T) -> bf16 WMMA B-fragment layout:
// out[((ntile*KT + ktile)*32 + lane)*16 + i] = bf16(W[n,k])
//   n = ntile*16 + (lane&15); k = ktile*32 + ((lane&16)?8:0) + ((i<8)? i : i+8)
// ---------------------------------------------------------------------------
__global__ void conv_w_bfrag(const float* __restrict__ W, unsigned short* __restrict__ out,
                             int N, int K) {
  long long idx = (long long)blockIdx.x * 256 + threadIdx.x;
  long long tot = (long long)N * K;
  if (idx >= tot) return;
  int i    = (int)(idx & 15);
  int lane = (int)((idx >> 4) & 31);
  long long f = idx >> 9;
  int KT = K >> 5;
  int kt = (int)(f % KT);
  int nt = (int)(f / KT);
  int n = nt * 16 + (lane & 15);
  int k = kt * 32 + ((lane & 16) ? 8 : 0) + ((i < 8) ? i : i + 8);
  out[idx] = f2bf(W[(long long)n * K + k]);
}

// ---------------------------------------------------------------------------
// src [B=64, T=256, IN=256] f32 -> per-t bf16 A-fragment layout:
// out[t][mtile(4)][ktile(8)][lane(32)][i(16)]
//   m = mtile*16 + (lane&15)
//   kk = ((lane&16)?8:0) + ((i<8)? i : (i-8)+16); k = ktile*32 + kk
// ---------------------------------------------------------------------------
__global__ void conv_src_afrag(const float* __restrict__ src, unsigned short* __restrict__ out) {
  long long idx = (long long)blockIdx.x * 256 + threadIdx.x;
  if (idx >= (long long)TENC * BATCH * INP) return;
  int i    = (int)(idx & 15);
  int lane = (int)((idx >> 4) & 31);
  long long f = idx >> 9;
  int kt = (int)(f & 7);
  int mt = (int)((f >> 3) & 3);
  int t  = (int)(f >> 5);
  int m  = mt * 16 + (lane & 15);
  int base = (lane & 16) ? 8 : 0;
  int kk = base + ((i < 8) ? i : (i - 8) + 16);
  int k  = kt * 32 + kk;
  out[idx] = f2bf(src[((long long)m * TENC + t) * INP + k]);
}

// ---------------------------------------------------------------------------
// Issue the async global->LDS copies for one staged K-group:
// A-frags for all 4 mtiles + B-frags for 3 gates x 2 local ntiles = 40KB.
// 256 threads x 10 x 16B issues, no VGPR data staging (DMA writes LDS).
// ---------------------------------------------------------------------------
__device__ __forceinline__ void stage_group_async(
    const unsigned short* __restrict__ aSrc,   // A-frags [4][KT][512]
    const unsigned short* __restrict__ bSrc,   // B-frags [192][KT][512]
    int KT, int nt0, int tid, int g,
    unsigned short* ldsbuf)                    // destination buffer in LDS
{
#pragma unroll
  for (int j = 0; j < 10; ++j) {
    int u = tid + 256 * j;                     // 16B unit index 0..2559
    const unsigned short* gp;
    if (u < 1024) {                            // A region: [mt(4)][ktl(4)][64 units]
      int mt  = u >> 8;
      int ktl = (u >> 6) & 3;
      int w   = u & 63;
      gp = aSrc + ((long long)(mt * KT + g * GROUP + ktl) * 512 + w * 8);
    } else {                                   // B region: [s(6)][ktl(4)][64 units]
      int v    = u - 1024;
      int s    = v >> 8;                       // s = gate*2 + tn
      int ktl  = (v >> 6) & 3;
      int w    = v & 63;
      int gate = s >> 1, tn = s & 1;
      gp = bSrc + ((long long)((nt0 + tn + gate * 64) * KT + g * GROUP + ktl) * 512 + w * 8);
    }
    async_g2l_b128(ldsbuf + u * 8, gp);
  }
}

// ---------------------------------------------------------------------------
// One GEMM phase of the GRU step, LDS double-buffered via async DMA:
//   wait asynccnt==0 ; barrier          -> group g resident in buf[g&1]
//   issue async copies of group g+1     -> buf[(g+1)&1] (readers already past)
//   compute group g: ds_load frags, 3 WMMAs per K-chunk
// Every fragment is fetched from L2 exactly once per block (4x less B traffic
// than the register-tiled version) and the DMA overlaps the WMMAs.
// ---------------------------------------------------------------------------
__device__ __forceinline__ void gemm_phase(
    const unsigned short* __restrict__ aSrc,   // A-frags [4][KT][512]
    const unsigned short* __restrict__ bSrc,   // B-frags [192][KT][512]
    int KT, int nt0, int tid, int lane, int wm, int wn,
    unsigned short* smem,                      // [2][BUFSZ]
    v8f& ar, v8f& az, v8f& a3)
{
  const int ngroups = KT / GROUP;

  stage_group_async(aSrc, bSrc, KT, nt0, tid, 0, smem);
  for (int g = 0; g < ngroups; ++g) {
    unsigned short* buf = smem + (g & 1) * BUFSZ;
    wait_async0();                             // this wave's copies of group g done
    __syncthreads();                           // whole block's group g visible
    if (g + 1 < ngroups)
      stage_group_async(aSrc, bSrc, KT, nt0, tid, g + 1,
                        smem + ((g + 1) & 1) * BUFSZ);
#pragma unroll
    for (int ktl = 0; ktl < GROUP; ++ktl) {
      v16bf a  = *(const v16bf*)(buf + (wm * GROUP + ktl) * 512 + lane * 16);
      v16bf fr = *(const v16bf*)(buf + ABUF + ((0 + wn) * GROUP + ktl) * 512 + lane * 16);
      v16bf fz = *(const v16bf*)(buf + ABUF + ((2 + wn) * GROUP + ktl) * 512 + lane * 16);
      v16bf fn = *(const v16bf*)(buf + ABUF + ((4 + wn) * GROUP + ktl) * 512 + lane * 16);
      ar = __builtin_amdgcn_wmma_f32_16x16x32_bf16(false, a, false, fr, (short)0, ar, false, false);
      az = __builtin_amdgcn_wmma_f32_16x16x32_bf16(false, a, false, fz, (short)0, az, false, false);
      a3 = __builtin_amdgcn_wmma_f32_16x16x32_bf16(false, a, false, fn, (short)0, a3, false, false);
    }
  }
  __syncthreads();                             // phase boundary: all reads done
}

// ---------------------------------------------------------------------------
// Fused GRU step: gates = WMMA GEMMs over [x | h], then sigmoid/tanh fusion.
// Grid: 32 blocks x 256 threads (8 waves). Wave (wm,wn) owns the 16x16 tile
// at rows wm*16.., cols (blockIdx.x*2+wn)*16.. of the [64 x 1024] hidden.
// Accumulators: r = i_r+h_r, z = i_z+h_z, i_n (x-part only), h_n (h-part only)
// (the n-gate keeps x/h parts separate because r multiplies only h_n).
// ---------------------------------------------------------------------------
__launch_bounds__(256)
__global__ void gru_step_kernel(
    const unsigned short* __restrict__ xfrag,  // A-frags of x_t  [4][dxT][512]
    int dxT,                                   // K_x / 32 (8 or 32)
    const unsigned short* __restrict__ hfrag,  // A-frags of h_{t-1} [4][32][512]
    const unsigned short* __restrict__ wiF,    // B-frags of Wi [192][dxT][512]
    const unsigned short* __restrict__ whF,    // B-frags of Wh [192][32][512]
    const float* __restrict__ bi, const float* __restrict__ bh,  // [3072]
    const float* __restrict__ hprev,           // [64][1024] f32
    float* __restrict__ hnew,                  // [64][1024] f32
    unsigned short* __restrict__ hnfrag,       // bf16 A-frags of h_t [4][32][512]
    unsigned short* __restrict__ xnfrag,       // same content for next layer (or null)
    float* __restrict__ seqout,                // optional f32 rows (or null)
    int seq_stride)
{
  __shared__ __align__(16) unsigned short smem[2 * BUFSZ];   // 80KB
  const int tid  = threadIdx.x;
  const int lane = tid & 31;
  const int wv   = tid >> 5;
  const int wm   = wv >> 1;            // 0..3  (M tile)
  const int wn   = wv & 1;             // 0..1
  const int nt0  = blockIdx.x * 2;
  const int nt   = nt0 + wn;           // hidden-col tile 0..63

  v8f ar = {};  // r gate (x+h)
  v8f az = {};  // z gate (x+h)
  v8f an = {};  // i_n (x only)
  v8f ah = {};  // h_n (h only)

  gemm_phase(xfrag, wiF, dxT, nt0, tid, lane, wm, wn, smem, ar, az, an);
  gemm_phase(hfrag, whF, 32,  nt0, tid, lane, wm, wn, smem, ar, az, ah);

  // ---- gate fusion (fp32) ----
  const int nl = lane & 15;
  const int n  = nt * 16 + nl;                 // hidden column 0..1023
  const float brz = bi[n]        + bh[n];
  const float bzz = bi[n + 1024] + bh[n + 1024];
  const float bin = bi[n + 2048];
  const float bhn = bh[n + 2048];
  const int mhi = (lane & 16) ? 8 : 0;

#pragma unroll
  for (int e = 0; e < 8; ++e) {
    const int m = wm * 16 + e + mhi;           // batch row 0..63
    float r  = sigm(ar[e] + brz);
    float z  = sigm(az[e] + bzz);
    float ng = tanhf(an[e] + bin + r * (ah[e] + bhn));
    float hp = hprev[m * HID + n];
    float h  = (1.0f - z) * ng + z * hp;

    hnew[m * HID + n] = h;
    if (seqout) seqout[(long long)m * seq_stride + n] = h;

    // scatter into bf16 A-fragment layout (K = 1024 -> 32 ktiles)
    unsigned short hb = f2bf(h);
    int kk = n & 31, ktile = n >> 5, mtile = m >> 4;
    int lp = (m & 15) + ((kk & 8) ? 16 : 0);
    int ip = (kk & 7) + ((kk & 16) ? 8 : 0);
    long long off = (((long long)(mtile * 32 + ktile)) * 32 + lp) * 16 + ip;
    hnfrag[off] = hb;
    if (xnfrag) xnfrag[off] = hb;
  }
}

// ---------------------------------------------------------------------------
// Decoder FC: pred = h1 @ fcW.T + fc_b.  M=64, N=256, K=1024.
// Grid: 8 blocks x 256 threads. Writes f32 pred into d_out and bf16 A-frags
// (K=256 layout) as next decoder step input.
// ---------------------------------------------------------------------------
__launch_bounds__(256)
__global__ void fc_step_kernel(
    const unsigned short* __restrict__ hfrag,  // [4][32][512]
    const unsigned short* __restrict__ wF,     // [16][32][512]
    const float* __restrict__ fcb,             // [256]
    float* __restrict__ outp, int out_stride,  // out[b, j] at outp + b*stride + j
    unsigned short* __restrict__ pfrag)        // [4][8][512]
{
  const int lane = threadIdx.x & 31;
  const int wv = threadIdx.x >> 5;
  const int wm = wv >> 1;
  const int wn = wv & 1;
  const int nt = blockIdx.x * 2 + wn;          // 0..15

  v8f acc = {};
  const unsigned short* ax = hfrag + ((long long)wm * 32) * 512 + lane * 16;
  const unsigned short* bw = wF + ((long long)nt * 32) * 512 + lane * 16;
  for (int kt = 0; kt < 32; ++kt) {
    v16bf a = *(const v16bf*)ax; ax += 512;
    v16bf b = *(const v16bf*)bw; bw += 512;
    if (kt + 4 < 32) __builtin_prefetch(bw + 4 * 512, 0, 0);
    acc = __builtin_amdgcn_wmma_f32_16x16x32_bf16(false, a, false, b, (short)0, acc, false, false);
  }

  const int j = nt * 16 + (lane & 15);         // output column 0..255
  const float bb = fcb[j];
  const int mhi = (lane & 16) ? 8 : 0;
#pragma unroll
  for (int e = 0; e < 8; ++e) {
    const int m = wm * 16 + e + mhi;
    float pv = acc[e] + bb;
    outp[(long long)m * out_stride + j] = pv;
    // bf16 A-frag for next decoder step (K = 256 -> 8 ktiles)
    int kk = j & 31, ktile = j >> 5, mtile = m >> 4;
    int lp = (m & 15) + ((kk & 8) ? 16 : 0);
    int ip = (kk & 7) + ((kk & 16) ? 8 : 0);
    pfrag[(((long long)(mtile * 8 + ktile)) * 32 + lp) * 16 + ip] = f2bf(pv);
  }
}

// ---------------------------------------------------------------------------
extern "C" void kernel_launch(void* const* d_in, const int* in_sizes, int n_in,
                              void* d_out, int out_size, void* d_ws, size_t ws_size,
                              hipStream_t stream) {
  (void)in_sizes; (void)n_in; (void)out_size; (void)ws_size;

  const float* src  = (const float*)d_in[0];
  // d_in[1] = trg (only its shape matters; teacher_forcing_ratio = 0)
  const float* eWi0 = (const float*)d_in[2];
  const float* eWh0 = (const float*)d_in[3];
  const float* eBi0 = (const float*)d_in[4];
  const float* eBh0 = (const float*)d_in[5];
  const float* eWi1 = (const float*)d_in[6];
  const float* eWh1 = (const float*)d_in[7];
  const float* eBi1 = (const float*)d_in[8];
  const float* eBh1 = (const float*)d_in[9];
  const float* dWi0 = (const float*)d_in[10];
  const float* dWh0 = (const float*)d_in[11];
  const float* dBi0 = (const float*)d_in[12];
  const float* dBh0 = (const float*)d_in[13];
  const float* dWi1 = (const float*)d_in[14];
  const float* dWh1 = (const float*)d_in[15];
  const float* dBi1 = (const float*)d_in[16];
  const float* dBh1 = (const float*)d_in[17];
  const float* fcW  = (const float*)d_in[18];
  const float* fcb  = (const float*)d_in[19];
  float* out = (float*)d_out;

  // d_out layout: out[64,64,256] | enc_state[64,256,1024] | dec_state[64,64,1024]
  const long long OUT_ENC = (long long)BATCH * TDEC * INP;              // 1,048,576
  const long long OUT_DEC = OUT_ENC + (long long)BATCH * TENC * HID;    // 17,825,792

  // ---- carve workspace (all 256B aligned) ----
  char* p = (char*)d_ws;
  auto take = [&](size_t bytes) -> char* {
    char* r = p; p += (bytes + 255) & ~(size_t)255; return r;
  };
  unsigned short* F_eWi0 = (unsigned short*)take((size_t)G3 * INP * 2);
  unsigned short* F_eWh0 = (unsigned short*)take((size_t)G3 * HID * 2);
  unsigned short* F_eWi1 = (unsigned short*)take((size_t)G3 * HID * 2);
  unsigned short* F_eWh1 = (unsigned short*)take((size_t)G3 * HID * 2);
  unsigned short* F_dWi0 = (unsigned short*)take((size_t)G3 * INP * 2);
  unsigned short* F_dWh0 = (unsigned short*)take((size_t)G3 * HID * 2);
  unsigned short* F_dWi1 = (unsigned short*)take((size_t)G3 * HID * 2);
  unsigned short* F_dWh1 = (unsigned short*)take((size_t)G3 * HID * 2);
  unsigned short* F_fcW  = (unsigned short*)take((size_t)INP * HID * 2);
  unsigned short* X0A    = (unsigned short*)take((size_t)TENC * BATCH * INP * 2); // src A-frags
  unsigned short* X1A    = (unsigned short*)take((size_t)BATCH * HID * 2);        // layer0->layer1 slot
  unsigned short* predA  = (unsigned short*)take((size_t)BATCH * INP * 2);        // decoder input
  float* h0f[2]; float* h1f[2]; unsigned short* h0A[2]; unsigned short* h1A[2];
  for (int i = 0; i < 2; ++i) {
    h0f[i] = (float*)take((size_t)BATCH * HID * 4);
    h1f[i] = (float*)take((size_t)BATCH * HID * 4);
    h0A[i] = (unsigned short*)take((size_t)BATCH * HID * 2);
    h1A[i] = (unsigned short*)take((size_t)BATCH * HID * 2);
  }

  // ---- one-time (per launch) packing: weights -> bf16 B-frags, src -> A-frags
  auto convW = [&](const float* W, unsigned short* o, int N, int K) {
    long long tot = (long long)N * K;
    conv_w_bfrag<<<(unsigned)((tot + 255) / 256), 256, 0, stream>>>(W, o, N, K);
  };
  convW(eWi0, F_eWi0, G3, INP);
  convW(eWh0, F_eWh0, G3, HID);
  convW(eWi1, F_eWi1, G3, HID);
  convW(eWh1, F_eWh1, G3, HID);
  convW(dWi0, F_dWi0, G3, INP);
  convW(dWh0, F_dWh0, G3, HID);
  convW(dWi1, F_dWi1, G3, HID);
  convW(dWh1, F_dWh1, G3, HID);
  convW(fcW,  F_fcW,  INP, HID);
  {
    long long tot = (long long)TENC * BATCH * INP;
    conv_src_afrag<<<(unsigned)((tot + 255) / 256), 256, 0, stream>>>(src, X0A);
  }

  // ---- zero initial states (h0 = 0, decoder input0 = 0) ----
  hipMemsetAsync(h0f[0], 0, (size_t)BATCH * HID * 4, stream);
  hipMemsetAsync(h1f[0], 0, (size_t)BATCH * HID * 4, stream);
  hipMemsetAsync(h0A[0], 0, (size_t)BATCH * HID * 2, stream);
  hipMemsetAsync(h1A[0], 0, (size_t)BATCH * HID * 2, stream);
  hipMemsetAsync(predA,  0, (size_t)BATCH * INP * 2, stream);

  // ---- encoder: per-t fused steps for layer0 then layer1 (wavefront order) ----
  for (int t = 0; t < TENC; ++t) {
    const int r = t & 1, w = r ^ 1;
    gru_step_kernel<<<32, 256, 0, stream>>>(
        X0A + (long long)t * BATCH * INP, /*dxT=*/INP / 32,
        h0A[r], F_eWi0, F_eWh0, eBi0, eBh0,
        h0f[r], h0f[w], h0A[w], X1A, nullptr, 0);
    gru_step_kernel<<<32, 256, 0, stream>>>(
        X1A, /*dxT=*/HID / 32,
        h1A[r], F_eWi1, F_eWh1, eBi1, eBh1,
        h1f[r], h1f[w], h1A[w], nullptr,
        out + OUT_ENC + (long long)t * HID, TENC * HID);  // enc_state[b,t,:]
  }

  // ---- decoder: autoregressive, parity continues (enc final hidden -> buf[0]) ----
  for (int t = 0; t < TDEC; ++t) {
    const int r = t & 1, w = r ^ 1;
    gru_step_kernel<<<32, 256, 0, stream>>>(
        predA, /*dxT=*/INP / 32,
        h0A[r], F_dWi0, F_dWh0, dBi0, dBh0,
        h0f[r], h0f[w], h0A[w], X1A, nullptr, 0);
    gru_step_kernel<<<32, 256, 0, stream>>>(
        X1A, /*dxT=*/HID / 32,
        h1A[r], F_dWi1, F_dWh1, dBi1, dBh1,
        h1f[r], h1f[w], h1A[w], nullptr,
        out + OUT_DEC + (long long)t * HID, TDEC * HID);  // dec_state[b,t,:]
    fc_step_kernel<<<8, 256, 0, stream>>>(
        h1A[w], F_fcW, fcb,
        out + (long long)t * INP, TDEC * INP,             // out[b,t,:]
        predA);
  }
}
